// GCN_16449724744842
// MI455X (gfx1250) — matrix operand order repaired and checked by reading, not compiled, not run on hardware
//
#include <hip/hip_runtime.h>

typedef __attribute__((ext_vector_type(16))) _Float16 v16h;
typedef __attribute__((ext_vector_type(8)))  float    v8f;
typedef __attribute__((ext_vector_type(4)))  int      v4i;

typedef __attribute__((address_space(1))) v4i* as1_v4i_p;   // global int4*
typedef __attribute__((address_space(3))) v4i* as3_v4i_p;   // LDS int4*

// ---------------------------------------------------------------------------
// Async global -> LDS staging (gfx1250 GLOBAL_LOAD_ASYNC_TO_LDS_B128 path,
// tracked by ASYNCcnt). Fallback: plain 16B copies through VGPRs.
// Must be called by ALL threads of the block before any early return; leaves
// LDS contents visible to the whole block (wait + barrier inside).
// ---------------------------------------------------------------------------
__device__ __forceinline__ void stage_to_lds(const void* g, void* l, int nbytes) {
#if __has_builtin(__builtin_amdgcn_global_load_async_to_lds_b128)
    for (int off = threadIdx.x * 16; off < nbytes; off += blockDim.x * 16) {
        __builtin_amdgcn_global_load_async_to_lds_b128(
            (as1_v4i_p)((char*)g + off), (as3_v4i_p)((char*)l + off), 0, 0);
    }
#if __has_builtin(__builtin_amdgcn_s_wait_asynccnt)
    __builtin_amdgcn_s_wait_asynccnt(0);
#else
    asm volatile("s_wait_asynccnt 0x0" ::: "memory");
#endif
#else
    const int4* gs = (const int4*)g;
    int4*       ls = (int4*)l;
    for (int i = threadIdx.x; i * 16 < nbytes; i += blockDim.x)
        ls[i] = gs[i];
#endif
    __syncthreads();
}

// ---------------------------------------------------------------------------
// Utility kernels
// ---------------------------------------------------------------------------
__global__ void fill0_kernel(float* __restrict__ p, long n) {
    long i = (long)blockIdx.x * blockDim.x + threadIdx.x;
    if (i < n) p[i] = 0.0f;
}

__global__ void degree_kernel(const int* __restrict__ src,
                              const int* __restrict__ dst,
                              float* __restrict__ deg_out,
                              float* __restrict__ deg_in, int E) {
    int e = blockIdx.x * blockDim.x + threadIdx.x;
    if (e >= E) return;
    unsafeAtomicAdd(&deg_out[src[e]], 1.0f);   // GLOBAL_ATOMIC_ADD_F32
    unsafeAtomicAdd(&deg_in[dst[e]],  1.0f);
}

__global__ void rsqrt_kernel(float* __restrict__ deg_out,
                             float* __restrict__ deg_in, int n) {
    int i = blockIdx.x * blockDim.x + threadIdx.x;
    if (i >= n) return;
    deg_out[i] = rsqrtf(fmaxf(deg_out[i], 1.0f));
    deg_in[i]  = rsqrtf(fmaxf(deg_in[i],  1.0f));
}

// ---------------------------------------------------------------------------
// Weight pre-pack: W[K,32] f32 row-major -> f16 blob in WMMA B-fragment order.
// Per 32-K chunk: 32 lanes x 32 halves; halves 0-15 = b0 (cols 0-15),
// halves 16-31 = b1 (cols 16-31); kb = chunk*32 + (lane>>4)*16 + (h&15).
// Lane then loads its whole fragment pair as 64 contiguous bytes from LDS.
// ---------------------------------------------------------------------------
__global__ void pack_w32_kernel(const float* __restrict__ W, int K,
                                _Float16* __restrict__ P) {
    int idx = blockIdx.x * blockDim.x + threadIdx.x;
    if (idx >= K * 32) return;
    int c = idx >> 10;                 // 32-K chunk
    int r = idx & 1023;
    int L = r >> 5, h = r & 31;        // lane, half
    int kb  = c * 32 + ((L >> 4) << 4) + (h & 15);
    int col = (L & 15) + ((h >> 4) << 4);
    P[idx] = (_Float16)W[kb * 32 + col];
}

// W3[32,8] -> b0-only fragments, cols 8-15 zero-padded. 512 halves.
__global__ void pack_w8_kernel(const float* __restrict__ W,
                               _Float16* __restrict__ P) {
    int idx = blockIdx.x * blockDim.x + threadIdx.x;
    if (idx >= 512) return;
    int L = idx >> 4, h = idx & 15;
    int kb  = ((L >> 4) << 4) + h;
    int col = L & 15;
    P[idx] = (col < 8) ? (_Float16)W[kb * 8 + col] : (_Float16)0.0f;
}

// ---------------------------------------------------------------------------
// WMMA GEMM: T[N,32] = (X[N,K] * norm_out[:,None]) @ W[K,32]
// One wave computes a 16-row x 32-col tile via two f16->f32 WMMAs per K-chunk.
// Wp = pre-packed f16 fragment blob (K*32 halves), async-staged into LDS.
// A layout (16-bit A 16x32): k = hi*16 + e (contiguous, same k order as packed B)
// D layout (32-bit C/D 16x16): vgpr r -> row r + 8*hi, col = lane&15
// ---------------------------------------------------------------------------
__global__ void gcn_gemm32(const float* __restrict__ X, int K,
                           const _Float16* __restrict__ Wp,
                           const float* __restrict__ norm_out,
                           float* __restrict__ T, int ntiles, int Nn) {
    extern __shared__ _Float16 lds_w[];            // K*32 halves, frag-ordered
    stage_to_lds(Wp, lds_w, K * 32 * (int)sizeof(_Float16));

    int wave = (int)(((long)blockIdx.x * blockDim.x + threadIdx.x) >> 5);
    int lane = threadIdx.x & 31;
    if (wave >= ntiles) return;                    // uniform per wave: EXEC full
    int row0 = wave * 16;
    int m    = lane & 15;                          // A row-in-tile / D column
    int hi   = lane >> 4;
    int row  = row0 + m; if (row >= Nn) row = Nn - 1;     // tail-safe load row
    float scale = norm_out[row];
    const float* xrow = X + (long)row * K;
    const v16h*  bfr  = (const v16h*)lds_w + (lane << 1); // lane's b0/b1 pair

    v8f acc0 = {}, acc1 = {};
    int nchunks = K >> 5;
    for (int ci = 0; ci < nchunks; ++ci) {
        int c = ci << 5;
        v16h a;
#pragma unroll
        for (int e = 0; e < 16; ++e) {
            int ka = c + (hi << 4) + e;
            a[e] = (_Float16)(xrow[ka] * scale);
        }
        v16h b0 = bfr[(ci << 6) + 0];              // ds_load_b128 x2
        v16h b1 = bfr[(ci << 6) + 1];
        acc0 = __builtin_amdgcn_wmma_f32_16x16x32_f16(false, a, false, b0,
                                                      (short)0, acc0, false, false);
        acc1 = __builtin_amdgcn_wmma_f32_16x16x32_f16(false, a, false, b1,
                                                      (short)0, acc1, false, false);
    }

    float* trow = T + (long)row0 * 32;
    if (row0 + 16 <= Nn) {                         // full tile: branch-free
#pragma unroll
        for (int r = 0; r < 8; ++r) {
            int o = (r + hi * 8) * 32 + m;
            trow[o]      = acc0[r];
            trow[o + 16] = acc1[r];
        }
    } else {
#pragma unroll
        for (int r = 0; r < 8; ++r) {
            int orow = row0 + r + hi * 8;
            if (orow < Nn) {
                T[(long)orow * 32 + m]      = acc0[r];
                T[(long)orow * 32 + 16 + m] = acc1[r];
            }
        }
    }
}

// T[N,8] = (X[N,32]*norm_out) @ W3[32,8]  (b0-only, cols 8-15 padded)
__global__ void gcn_gemm8(const float* __restrict__ X,
                          const _Float16* __restrict__ Wp,
                          const float* __restrict__ norm_out,
                          float* __restrict__ T, int ntiles, int Nn) {
    __shared__ _Float16 lds_w[512];                // frag-ordered
    stage_to_lds(Wp, lds_w, 512 * (int)sizeof(_Float16));

    int wave = (int)(((long)blockIdx.x * blockDim.x + threadIdx.x) >> 5);
    int lane = threadIdx.x & 31;
    if (wave >= ntiles) return;
    int row0 = wave * 16;
    int m    = lane & 15;
    int hi   = lane >> 4;
    int row  = row0 + m; if (row >= Nn) row = Nn - 1;
    float scale = norm_out[row];
    const float* xrow = X + (long)row * 32;

    v16h a;
#pragma unroll
    for (int e = 0; e < 16; ++e) {
        int ka = (hi << 4) + e;
        a[e] = (_Float16)(xrow[ka] * scale);
    }
    v16h b0 = ((const v16h*)lds_w)[lane];
    v8f acc = {};
    acc = __builtin_amdgcn_wmma_f32_16x16x32_f16(false, a, false, b0,
                                                 (short)0, acc, false, false);

    if (row0 + 16 <= Nn) {
        if (m < 8) {
            float* trow = T + (long)row0 * 8;
#pragma unroll
            for (int r = 0; r < 8; ++r)
                trow[(r + hi * 8) * 8 + m] = acc[r];
        }
    } else {
#pragma unroll
        for (int r = 0; r < 8; ++r) {
            int orow = row0 + r + hi * 8;
            if (m < 8 && orow < Nn) T[(long)orow * 8 + m] = acc[r];
        }
    }
}

// ---------------------------------------------------------------------------
// Edge scatter:  AGG[dst[e], j] += T[src[e], j]   (one lane per (edge, chan))
// ---------------------------------------------------------------------------
template <int LOGW>
__global__ void scatter_add_kernel(const int* __restrict__ src,
                                   const int* __restrict__ dst,
                                   const float* __restrict__ T,
                                   float* __restrict__ AGG, int E) {
    long gid = (long)blockIdx.x * blockDim.x + threadIdx.x;
    if (gid >= ((long)E << LOGW)) return;
    int e = (int)(gid >> LOGW);
    int j = (int)(gid & ((1 << LOGW) - 1));
    int s = src[e], d = dst[e];
    unsafeAtomicAdd(&AGG[((long)d << LOGW) + j], T[((long)s << LOGW) + j]);
}

// ---------------------------------------------------------------------------
// Epilogues
// ---------------------------------------------------------------------------
__global__ void bias_relu32_kernel(const float* __restrict__ AGG,
                                   const float* __restrict__ norm_in,
                                   const float* __restrict__ b,
                                   float* __restrict__ H, int Nn) {
    long gid = (long)blockIdx.x * blockDim.x + threadIdx.x;
    if (gid >= (long)Nn * 32) return;
    int i = (int)(gid >> 5), j = (int)(gid & 31);
    H[gid] = fmaxf(AGG[gid] * norm_in[i] + b[j], 0.0f);
}

__global__ void bias8_kernel(float* __restrict__ OUT,
                             const float* __restrict__ norm_in,
                             const float* __restrict__ b, int Nn) {
    long gid = (long)blockIdx.x * blockDim.x + threadIdx.x;
    if (gid >= (long)Nn * 8) return;
    int i = (int)(gid >> 3), j = (int)(gid & 7);
    OUT[gid] = OUT[gid] * norm_in[i] + b[j];   // no relu on last layer
}

// ---------------------------------------------------------------------------
// Launcher
// ---------------------------------------------------------------------------
static inline int cdiv(long a, long b) { return (int)((a + b - 1) / b); }

extern "C" void kernel_launch(void* const* d_in, const int* in_sizes, int n_in,
                              void* d_out, int out_size, void* d_ws, size_t ws_size,
                              hipStream_t stream) {
    const float* features = (const float*)d_in[0];
    const int*   src      = (const int*)d_in[1];
    const int*   dst      = (const int*)d_in[2];
    const float* W0 = (const float*)d_in[3];
    const float* b0 = (const float*)d_in[4];
    const float* W1 = (const float*)d_in[5];
    const float* b1 = (const float*)d_in[6];
    const float* W2 = (const float*)d_in[7];
    const float* b2 = (const float*)d_in[8];
    const float* W3 = (const float*)d_in[9];
    const float* b3 = (const float*)d_in[10];
    float*       out = (float*)d_out;

    const int Nn = in_sizes[0] / 128;   // 100000
    const int E  = in_sizes[1];         // 1600000

    // workspace: norm_out[N] | norm_in[N] | bufA[N*32] | bufB[N*32]
    //            | packed f16 weights (~13 KB)           (~26.4 MB total)
    float* ws       = (float*)d_ws;
    float* norm_out = ws;
    float* norm_in  = ws + Nn;
    float* bufA     = ws + 2 * (size_t)Nn;
    float* bufB     = bufA + (size_t)Nn * 32;
    _Float16* P0 = (_Float16*)(bufB + (size_t)Nn * 32);   // 4096 halves
    _Float16* P1 = P0 + 4096;                             // 1024
    _Float16* P2 = P1 + 1024;                             // 1024
    _Float16* P3 = P2 + 1024;                             // 512

    const int B = 256;
    const int ntiles  = (Nn + 15) / 16;
    const int gblocks = (ntiles + 7) / 8;          // 8 waves per block
    const long n32 = (long)Nn * 32;
    const size_t lds128 = 128 * 32 * sizeof(_Float16);  // 8 KB
    const size_t lds32  =  32 * 32 * sizeof(_Float16);  // 2 KB

    // degree norms + weight pre-pack (tiny)
    fill0_kernel<<<cdiv(2L * Nn, B), B, 0, stream>>>(norm_out, 2L * Nn);
    degree_kernel<<<cdiv(E, B), B, 0, stream>>>(src, dst, norm_out, norm_in, E);
    rsqrt_kernel<<<cdiv(Nn, B), B, 0, stream>>>(norm_out, norm_in, Nn);
    pack_w32_kernel<<<cdiv(4096, B), B, 0, stream>>>(W0, 128, P0);
    pack_w32_kernel<<<cdiv(1024, B), B, 0, stream>>>(W1, 32, P1);
    pack_w32_kernel<<<cdiv(1024, B), B, 0, stream>>>(W2, 32, P2);
    pack_w8_kernel<<<cdiv(512, B), B, 0, stream>>>(W3, P3);

    // Layer 0: features[N,128] -> bufA -> scatter -> bufB -> relu -> bufA
    gcn_gemm32<<<gblocks, B, lds128, stream>>>(features, 128, P0, norm_out, bufA, ntiles, Nn);
    fill0_kernel<<<cdiv(n32, B), B, 0, stream>>>(bufB, n32);
    scatter_add_kernel<5><<<cdiv((long)E * 32, B), B, 0, stream>>>(src, dst, bufA, bufB, E);
    bias_relu32_kernel<<<cdiv(n32, B), B, 0, stream>>>(bufB, norm_in, b0, bufA, Nn);

    // Layer 1
    gcn_gemm32<<<gblocks, B, lds32, stream>>>(bufA, 32, P1, norm_out, bufB, ntiles, Nn);
    fill0_kernel<<<cdiv(n32, B), B, 0, stream>>>(bufA, n32);
    scatter_add_kernel<5><<<cdiv((long)E * 32, B), B, 0, stream>>>(src, dst, bufB, bufA, E);
    bias_relu32_kernel<<<cdiv(n32, B), B, 0, stream>>>(bufA, norm_in, b1, bufB, Nn);

    // Layer 2
    gcn_gemm32<<<gblocks, B, lds32, stream>>>(bufB, 32, P2, norm_out, bufA, ntiles, Nn);
    fill0_kernel<<<cdiv(n32, B), B, 0, stream>>>(bufB, n32);
    scatter_add_kernel<5><<<cdiv((long)E * 32, B), B, 0, stream>>>(src, dst, bufA, bufB, E);
    bias_relu32_kernel<<<cdiv(n32, B), B, 0, stream>>>(bufB, norm_in, b2, bufA, Nn);

    // Layer 3 (width 8, no relu): bufA -> bufB(N*8) -> scatter -> out -> bias
    gcn_gemm8<<<gblocks, B, 0, stream>>>(bufA, P3, norm_out, bufB, ntiles, Nn);
    fill0_kernel<<<cdiv((long)Nn * 8, B), B, 0, stream>>>(out, (long)Nn * 8);
    scatter_add_kernel<3><<<cdiv((long)E * 8, B), B, 0, stream>>>(src, dst, bufB, out, E);
    bias8_kernel<<<cdiv((long)Nn * 8, B), B, 0, stream>>>(out, norm_in, b3, Nn);
}
